// SerialBlock_45930380263595
// MI455X (gfx1250) — matrix operand order repaired and checked by reading, not compiled, not run on hardware
//
// CDNA5 / gfx1250 implementation of the CoaT serial block.
// GEMMs (qkv / proj / mlp1 / mlp2) run on v_wmma_f32_16x16x32_bf16 with
// LDS-staged A tiles (async global->LDS via gfx1250 builtin) and
// fragment-prepacked bf16 weights (L2-resident). Wave tile 64x64
// (16 WMMA / K-step per wave), workgroup tile 128x256 (2x4 waves).
#include <hip/hip_runtime.h>
#include <stdint.h>

#define Bb   8
#define Nn   1025
#define Cc   768
#define NHh  12
#define CHh  64
#define HIDh 3072
#define Mm   (Bb * Nn)      // 8200 rows
#define N3   (3 * Cc)       // 2304

typedef __bf16 v16bf __attribute__((ext_vector_type(16)));
typedef float  v8f   __attribute__((ext_vector_type(8)));
typedef int    v4i   __attribute__((ext_vector_type(4)));

#if defined(__has_builtin)
#if __has_builtin(__builtin_amdgcn_global_load_async_to_lds_b128) && \
    __has_builtin(__builtin_amdgcn_s_wait_asynccnt)
#define HAS_ASYNC_LDS 1
#endif
#endif

static __device__ __forceinline__ unsigned short f2bf(float f) {
    union { float f; unsigned u; } x; x.f = f;
    unsigned r = x.u + 0x7FFFu + ((x.u >> 16) & 1u);   // round-nearest-even
    return (unsigned short)(r >> 16);
}

static __device__ __forceinline__ void async_cp16(const void* g, void* l) {
#if HAS_ASYNC_LDS
    __builtin_amdgcn_global_load_async_to_lds_b128(
        (__attribute__((address_space(1))) v4i*)g,
        (__attribute__((address_space(3))) v4i*)l, 0, 0);
#else
    *(uint4*)l = *(const uint4*)g;
#endif
}
static __device__ __forceinline__ void wait_async() {
#if HAS_ASYNC_LDS
    __builtin_amdgcn_s_wait_asynccnt(0);
#endif
}

// ---------------------------------------------------------------- weight pack
// Pack f32 KxN row-major into fragment-native bf16:
//   dst[((tn*(K/32)+tk)*32 + 16*((k%32)/8%2) + (n%16))*16 + 8*((k%32)/16) + (k%8)]
// so a lane (n == lane%16, ISA table k order) loads one contiguous v16bf.
__global__ void pack_b_kernel(const float* __restrict__ src,
                              unsigned short* __restrict__ dst, int K, int N) {
    size_t i = (size_t)blockIdx.x * 256 + threadIdx.x;
    if (i >= (size_t)K * N) return;
    int k = (int)(i / N), n = (int)(i % N);
    size_t o = (((size_t)(n >> 4) * (K >> 5) + (k >> 5)) * 32 + ((k >> 3) & 1) * 16 + (n & 15)) * 16 + ((k >> 4) & 1) * 8 + (k & 7);
    dst[o] = f2bf(src[i]);
}

// ---------------------------------------------------------------- WMMA GEMM
// C[M,N] = A[M,K](bf16) * Bpacked[K,N](bf16) ; epilogue by mode:
//   0: out.f32 = acc + bias
//   1: out.f32 = acc + bias + resid
//   2: out.bf16 = gelu(acc + bias)
#define APAD 40   // ushort row pad: 80B rows -> 16B-aligned b128 LDS writes, no bank conflicts
__global__ __launch_bounds__(256) void gemm_bf16_kernel(
    const unsigned short* __restrict__ A, const unsigned short* __restrict__ Bp,
    const float* __restrict__ bias, const float* __restrict__ resid,
    void* __restrict__ out, int M, int K, int N, int mode) {
    __shared__ __attribute__((aligned(16))) unsigned short As[2][128][APAD];
    const int tid = threadIdx.x, lane = tid & 31, wid = tid >> 5;
    const int wm = wid & 1, wn = wid >> 1;           // 2x4 waves -> 128x256 tile
    const int bm = blockIdx.x, bn = blockIdx.y;
    const int kTiles = K >> 5;
    const v16bf* __restrict__ Bv = (const v16bf*)Bp;

    const int lr = tid >> 1, lc = (tid & 1) * 16;    // 2 threads per 32-col row
    auto stage = [&](int s, int kt) {
        int row = bm * 128 + lr; if (row >= M) row = M - 1;
        const unsigned short* gp = A + (size_t)row * K + (size_t)kt * 32 + lc;
        unsigned short* lp = &As[s][lr][lc];
        async_cp16(gp, lp);
        async_cp16(gp + 8, lp + 8);
    };

    v8f acc[4][4];
    #pragma unroll
    for (int i = 0; i < 4; ++i)
        #pragma unroll
        for (int j = 0; j < 4; ++j) acc[i][j] = (v8f){};

    stage(0, 0);
    for (int kt = 0; kt < kTiles; ++kt) {
        const int s = kt & 1;
        wait_async();
        __syncthreads();
        if (kt + 1 < kTiles) stage(s ^ 1, kt + 1);

        // A fragments from LDS (layout per ISA 16-bit A 16x32 table)
        union AU { unsigned u[8]; v16bf v; } a[4];
        const int h8 = (lane >> 4) * 8;
        #pragma unroll
        for (int mt = 0; mt < 4; ++mt) {
            const int m = wm * 64 + mt * 16 + (lane & 15);
            #pragma unroll
            for (int v = 0; v < 8; ++v) {
                const int kb = ((v < 4) ? 2 * v : 16 + 2 * (v - 4)) + h8;
                a[mt].u[v] = *(const unsigned*)&As[s][m][kb];
            }
        }
        // B fragments straight from packed global (L2-resident weights)
        v16bf bf[4];
        #pragma unroll
        for (int nt = 0; nt < 4; ++nt) {
            size_t idx = ((size_t)(bn * 16 + wn * 4 + nt) * kTiles + kt) * 32 + lane;
            bf[nt] = Bv[idx];
            if (kt + 1 < kTiles)
                __builtin_prefetch(&Bv[idx + 32], 0, 1);
        }
        #pragma unroll
        for (int mt = 0; mt < 4; ++mt)
            #pragma unroll
            for (int nt = 0; nt < 4; ++nt)
                acc[mt][nt] = __builtin_amdgcn_wmma_f32_16x16x32_bf16(
                    false, a[mt].v, false, bf[nt], (short)0, acc[mt][nt], false, false);
    }

    // epilogue (C layout: VGPR g -> row g + 8*(lane>>4); col = lane&15)
    #pragma unroll
    for (int mt = 0; mt < 4; ++mt)
        #pragma unroll
        for (int nt = 0; nt < 4; ++nt) {
            const int col = bn * 256 + wn * 64 + nt * 16 + (lane & 15);
            const float bc = bias ? bias[col] : 0.f;
            #pragma unroll
            for (int g = 0; g < 8; ++g) {
                const int row = bm * 128 + wm * 64 + mt * 16 + g + 8 * (lane >> 4);
                if (row >= M) continue;
                const size_t o = (size_t)row * N + col;
                float v = acc[mt][nt][g] + bc;
                if (mode == 0) ((float*)out)[o] = v;
                else if (mode == 1) ((float*)out)[o] = v + resid[o];
                else {
                    v = 0.5f * v * (1.f + erff(v * 0.70710678118654752f));
                    ((unsigned short*)out)[o] = f2bf(v);
                }
            }
        }
}

// ---------------------------------------------------------------- CPE (3x3 dw + residual, copy cls)
__global__ void cpe_kernel(const float* __restrict__ x, const float* __restrict__ w,
                           const float* __restrict__ bias, float* __restrict__ xw) {
    const int bn = blockIdx.x, b = bn / Nn, n = bn % Nn, t = threadIdx.x;
    const size_t base = (size_t)bn * Cc;
    if (n == 0) {
        #pragma unroll
        for (int i = 0; i < 3; ++i) xw[base + t + i * 256] = x[base + t + i * 256];
        return;
    }
    const int p = n - 1, py = p >> 5, px = p & 31;
    #pragma unroll
    for (int i = 0; i < 3; ++i) {
        const int c = t + i * 256;
        float acc = bias[c];
        for (int ky = 0; ky < 3; ++ky) {
            const int iy = py + ky - 1; if (iy < 0 || iy > 31) continue;
            for (int kx = 0; kx < 3; ++kx) {
                const int ix = px + kx - 1; if (ix < 0 || ix > 31) continue;
                acc += w[(ky * 3 + kx) * Cc + c] *
                       x[((size_t)b * Nn + 1 + iy * 32 + ix) * Cc + c];
            }
        }
        xw[base + c] = acc + x[base + c];
    }
}

// ---------------------------------------------------------------- LayerNorm (f32 in -> bf16 out)
__global__ void ln_kernel(const float* __restrict__ x, const float* __restrict__ g,
                          const float* __restrict__ bt, unsigned short* __restrict__ out) {
    __shared__ float red[256];
    const int t = threadIdx.x;
    const size_t base = (size_t)blockIdx.x * Cc;
    float v0 = x[base + t], v1 = x[base + t + 256], v2 = x[base + t + 512];
    red[t] = v0 + v1 + v2; __syncthreads();
    for (int o = 128; o > 0; o >>= 1) { if (t < o) red[t] += red[t + o]; __syncthreads(); }
    const float mean = red[0] * (1.f / Cc); __syncthreads();
    const float d0 = v0 - mean, d1 = v1 - mean, d2 = v2 - mean;
    red[t] = d0 * d0 + d1 * d1 + d2 * d2; __syncthreads();
    for (int o = 128; o > 0; o >>= 1) { if (t < o) red[t] += red[t + o]; __syncthreads(); }
    const float rs = rsqrtf(red[0] * (1.f / Cc) + 1e-6f);
    out[base + t]       = f2bf(d0 * rs * g[t]       + bt[t]);
    out[base + t + 256] = f2bf(d1 * rs * g[t + 256] + bt[t + 256]);
    out[base + t + 512] = f2bf(d2 * rs * g[t + 512] + bt[t + 512]);
}

// ---------------------------------------------------------------- softmax stats over tokens (per b,h,c)
__global__ void ksm_stats_kernel(const float* __restrict__ qkv, float* __restrict__ stats) {
    const int bh = blockIdx.x, b = bh / NHh, h = bh % NHh, c = threadIdx.x;
    const float* kp = qkv + (size_t)b * Nn * N3 + Cc + h * CHh + c;
    float mx = -1e30f;
    for (int n = 0; n < Nn; ++n) mx = fmaxf(mx, kp[(size_t)n * N3]);
    float s = 0.f;
    for (int n = 0; n < Nn; ++n) s += expf(kp[(size_t)n * N3] - mx);
    stats[bh * 128 + c] = mx;
    stats[bh * 128 + 64 + c] = 1.f / s;
}

// ---------------------------------------------------------------- fa = softmax(k)^T @ v  (64x64 per b,h)
__global__ void fa_kernel(const float* __restrict__ qkv, const float* __restrict__ stats,
                          float* __restrict__ fa) {
    __shared__ float kv[64], vv[64];
    const int bh = blockIdx.x, b = bh / NHh, h = bh % NHh, t = threadIdx.x;
    float mx = 0.f, rs = 0.f;
    if (t < 64) { mx = stats[bh * 128 + t]; rs = stats[bh * 128 + 64 + t]; }
    float acc[16];
    #pragma unroll
    for (int j = 0; j < 16; ++j) acc[j] = 0.f;
    const float* base = qkv + (size_t)b * Nn * N3 + h * CHh;
    const int c = t & 63, d0 = (t >> 6) * 16;
    for (int n = 0; n < Nn; ++n) {
        if (t < 64)       kv[t]      = expf(base[(size_t)n * N3 + Cc + t] - mx) * rs;
        else if (t < 128) vv[t - 64] = base[(size_t)n * N3 + 2 * Cc + (t - 64)];
        __syncthreads();
        #pragma unroll
        for (int j = 0; j < 16; ++j) acc[j] += kv[c] * vv[d0 + j];
        __syncthreads();
    }
    float* fo = fa + ((size_t)bh * 64 + c) * 64 + d0;
    #pragma unroll
    for (int j = 0; j < 16; ++j) fo[j] = acc[j];
}

// ---------------------------------------------------------------- CRPE depthwise (3/5/7 per group)
__global__ void crpe_kernel(const float* __restrict__ qkv,
                            const float* __restrict__ w3, const float* __restrict__ b3,
                            const float* __restrict__ w5, const float* __restrict__ b5,
                            const float* __restrict__ w7, const float* __restrict__ b7,
                            float* __restrict__ cv) {
    const int bp = blockIdx.x, b = bp >> 10, p = bp & 1023;
    const int py = p >> 5, px = p & 31, t = threadIdx.x;
    #pragma unroll
    for (int i = 0; i < 3; ++i) {
        const int c = t + i * 256, g = c >> 8, cl = c & 255;
        const int ks = 3 + 2 * g, pad = g + 1;
        const float* w = (g == 0) ? w3 : (g == 1) ? w5 : w7;
        const float* bb = (g == 0) ? b3 : (g == 1) ? b5 : b7;
        float acc = bb[cl];
        for (int ky = 0; ky < ks; ++ky) {
            const int iy = py + ky - pad; if (iy < 0 || iy > 31) continue;
            for (int kx = 0; kx < ks; ++kx) {
                const int ix = px + kx - pad; if (ix < 0 || ix > 31) continue;
                acc += w[(ky * ks + kx) * 256 + cl] *
                       qkv[((size_t)b * Nn + 1 + iy * 32 + ix) * N3 + 2 * Cc + c];
            }
        }
        cv[(size_t)bp * Cc + c] = acc;
    }
}

// ---------------------------------------------------------------- out = SCALE*q@fa + q*cv  (bf16 for proj GEMM)
__global__ void attout_kernel(const float* __restrict__ qkv, const float* __restrict__ fa,
                              const float* __restrict__ cv, unsigned short* __restrict__ attn) {
    __shared__ float qs[Cc];
    const int bn = blockIdx.x, b = bn / Nn, n = bn % Nn, t = threadIdx.x;
    const size_t qbase = (size_t)bn * N3;
    #pragma unroll
    for (int i = 0; i < 3; ++i) qs[t + i * 256] = qkv[qbase + t + i * 256];
    __syncthreads();
    #pragma unroll
    for (int i = 0; i < 3; ++i) {
        const int c = t + i * 256, h = c >> 6, d = c & 63;
        const float* fah = fa + (size_t)(b * NHh + h) * 64 * 64 + d;
        const float* qh = qs + h * 64;
        float s = 0.f;
        #pragma unroll 8
        for (int k = 0; k < 64; ++k) s += qh[k] * fah[k * 64];
        float o = 0.125f * s;   // CH^-0.5
        if (n > 0) o += qs[c] * cv[((size_t)(b * 1024 + (n - 1))) * Cc + c];
        attn[(size_t)bn * Cc + c] = f2bf(o);
    }
}

// ---------------------------------------------------------------- launch
extern "C" void kernel_launch(void* const* d_in, const int* in_sizes, int n_in,
                              void* d_out, int out_size, void* d_ws, size_t ws_size,
                              hipStream_t stream) {
    const float* x      = (const float*)d_in[0];
    const float* cpe_w  = (const float*)d_in[3];
    const float* cpe_b  = (const float*)d_in[4];
    const float* g1     = (const float*)d_in[5];
    const float* bta1   = (const float*)d_in[6];
    const float* qkv_w  = (const float*)d_in[7];
    const float* qkv_b  = (const float*)d_in[8];
    const float* cw3    = (const float*)d_in[9];
    const float* cb3    = (const float*)d_in[10];
    const float* cw5    = (const float*)d_in[11];
    const float* cb5    = (const float*)d_in[12];
    const float* cw7    = (const float*)d_in[13];
    const float* cb7    = (const float*)d_in[14];
    const float* proj_w = (const float*)d_in[15];
    const float* proj_b = (const float*)d_in[16];
    const float* g2     = (const float*)d_in[17];
    const float* bta2   = (const float*)d_in[18];
    const float* mw1    = (const float*)d_in[19];
    const float* mb1    = (const float*)d_in[20];
    const float* mw2    = (const float*)d_in[21];
    const float* mb2    = (const float*)d_in[22];

    char* ws = (char*)d_ws;
    size_t off = 0;
    auto take = [&](size_t bytes) { size_t o = off; off = (off + bytes + 255) & ~(size_t)255; return o; };
    float*          x_work = (float*)(ws + take((size_t)Mm * Cc * 4));
    unsigned short* curA   = (unsigned short*)(ws + take((size_t)Mm * Cc * 2));
    float*          qkvb   = (float*)(ws + take((size_t)Mm * N3 * 4));
    unsigned short* wq     = (unsigned short*)(ws + take((size_t)Cc * N3 * 2));
    unsigned short* wp     = (unsigned short*)(ws + take((size_t)Cc * Cc * 2));
    unsigned short* w1     = (unsigned short*)(ws + take((size_t)Cc * HIDh * 2));
    unsigned short* w2     = (unsigned short*)(ws + take((size_t)HIDh * Cc * 2));
    float*          stats  = (float*)(ws + take((size_t)Bb * NHh * 128 * 4));
    float*          fab    = (float*)(ws + take((size_t)Bb * NHh * 64 * 64 * 4));
    float*          cvb    = (float*)(ws + take((size_t)Bb * 1024 * Cc * 4));
    unsigned short* attnb  = (unsigned short*)(ws + take((size_t)Mm * Cc * 2));
    float*          x2     = (float*)(ws + take((size_t)Mm * Cc * 4));
    unsigned short* curB   = (unsigned short*)(ws + take((size_t)Mm * Cc * 2));
    unsigned short* hid    = (unsigned short*)(ws + take((size_t)Mm * HIDh * 2));

    // weight packing (deterministic, every call)
    pack_b_kernel<<<(Cc * N3 + 255) / 256, 256, 0, stream>>>(qkv_w, wq, Cc, N3);
    pack_b_kernel<<<(Cc * Cc + 255) / 256, 256, 0, stream>>>(proj_w, wp, Cc, Cc);
    pack_b_kernel<<<(Cc * HIDh + 255) / 256, 256, 0, stream>>>(mw1, w1, Cc, HIDh);
    pack_b_kernel<<<(HIDh * Cc + 255) / 256, 256, 0, stream>>>(mw2, w2, HIDh, Cc);

    cpe_kernel<<<Mm, 256, 0, stream>>>(x, cpe_w, cpe_b, x_work);
    ln_kernel<<<Mm, 256, 0, stream>>>(x_work, g1, bta1, curA);

    dim3 gq((Mm + 127) / 128, N3 / 256);
    gemm_bf16_kernel<<<gq, 256, 0, stream>>>(curA, wq, qkv_b, nullptr, qkvb, Mm, Cc, N3, 0);

    ksm_stats_kernel<<<Bb * NHh, 64, 0, stream>>>(qkvb, stats);
    fa_kernel<<<Bb * NHh, 256, 0, stream>>>(qkvb, stats, fab);
    crpe_kernel<<<Bb * 1024, 256, 0, stream>>>(qkvb, cw3, cb3, cw5, cb5, cw7, cb7, cvb);
    attout_kernel<<<Mm, 256, 0, stream>>>(qkvb, fab, cvb, attnb);

    dim3 gp((Mm + 127) / 128, Cc / 256);
    gemm_bf16_kernel<<<gp, 256, 0, stream>>>(attnb, wp, proj_b, x_work, x2, Mm, Cc, Cc, 1);

    ln_kernel<<<Mm, 256, 0, stream>>>(x2, g2, bta2, curB);

    dim3 g1d((Mm + 127) / 128, HIDh / 256);
    gemm_bf16_kernel<<<g1d, 256, 0, stream>>>(curB, w1, mb1, nullptr, hid, Mm, Cc, HIDh, 2);

    dim3 g2d((Mm + 127) / 128, Cc / 256);
    gemm_bf16_kernel<<<g2d, 256, 0, stream>>>(hid, w2, mb2, x2, (float*)d_out, Mm, HIDh, Cc, 1);
}
